// unit_tcn_G_32014686224448
// MI455X (gfx1250) — matrix operand order repaired and checked by reading, not compiled, not run on hardware
//
#include <hip/hip_runtime.h>
#include <hip/hip_bf16.h>

// ---------------------------------------------------------------------------
// Types for CDNA5 WMMA (wave32): A/B = 16 bf16 per lane, C/D = 8 f32 per lane.
// ---------------------------------------------------------------------------
typedef __attribute__((ext_vector_type(16))) __bf16 v16bf;
typedef __attribute__((ext_vector_type(8)))  float  v8f;

__device__ __forceinline__ unsigned short f2bf(float f) {
  // round-to-nearest-even float32 -> bf16
  unsigned u = __float_as_uint(f);
  unsigned r = u + 0x7FFFu + ((u >> 16) & 1u);
  return (unsigned short)(r >> 16);
}

// ---------------------------------------------------------------------------
// Pack x (N,C,T,V) f32  ->  xT (N, T*V, C) bf16   (channel-last, WMMA-B ready)
// LDS transpose for coalesced read+write. grid=(125 tiles of 60 cols, N)
// ---------------------------------------------------------------------------
__global__ void __launch_bounds__(256) pack_x_kernel(const float* __restrict__ x,
                                                     unsigned short* __restrict__ xT) {
  __shared__ float tile[64][61];
  const int n = blockIdx.y;
  const int base = blockIdx.x * 60;                       // 7500 = 125 * 60
  const size_t xoff = (size_t)n * 64 * 7500;
  for (int e = threadIdx.x; e < 64 * 60; e += 256) {
    int c = e / 60, i = e - c * 60;
    tile[c][i] = x[xoff + (size_t)c * 7500 + base + i];
  }
  __syncthreads();
  const size_t ooff = (size_t)n * 7500 * 64;
  for (int e = threadIdx.x; e < 60 * 64; e += 256) {
    int i = e >> 6, c = e & 63;
    xT[ooff + (size_t)(base + i) * 64 + c] = f2bf(tile[c][i]);
  }
}

// ---------------------------------------------------------------------------
// Pack conv weights (OC,64,9,1) f32 -> bf16 fragments in a contiguous-k WMMA A order (B loads match):
//   dst[(((octile*9 + k)*2 + kc)*32 + lane)*16 + i]
//   lane<16 : row=octile*16+lane,   K offsets {0..15}
//   lane>=16: row=octile*16+lane-16,K offsets {16..31}
// Rows [0,rows1) come from w1, rows [rows1,OC) from w2 (conv1/conv2 fusion).
// ---------------------------------------------------------------------------
__global__ void __launch_bounds__(256) pack_w_kernel(const float* __restrict__ w1,
                                                     const float* __restrict__ w2,
                                                     int rows1, int OC,
                                                     unsigned short* __restrict__ dst) {
  int idx = blockIdx.x * 256 + threadIdx.x;
  if (idx >= OC * 576) return;
  int i    = idx & 15;
  int lane = (idx >> 4) & 31;
  int rest = idx >> 9;
  int kc   = rest & 1;  rest >>= 1;
  int k    = rest % 9;
  int octile = rest / 9;
  int half = lane >> 4;
  int oc   = octile * 16 + (lane & 15);
  int coff = i + (half << 4);
  int cc   = kc * 32 + coff;                              // channel index 0..63
  const float* w = (oc < rows1) ? w1 : w2;
  int row = (oc < rows1) ? oc : oc - rows1;
  dst[idx] = f2bf(w[(row * 64 + cc) * 9 + k]);
}

// ---------------------------------------------------------------------------
// Temporal conv (kernel 9, pad 4) as chained bf16 WMMAs with f32 accum.
// Each wave computes a 16(oc) x 32(col) tile: two N-tiles share every A
// fragment (weights stay in registers for both), 36 back-to-back WMMAs.
//   in : xT (N, 7500, 64) bf16   out: (N, OC, 7500) f32 (+ bias)
// grid = (235 col-pairs, OC/64, N), block = 128 (4 waves = 4 oc-tiles)
// ---------------------------------------------------------------------------
__global__ void __launch_bounds__(128) tconv_wmma_kernel(
    const unsigned short* __restrict__ xT, const unsigned short* __restrict__ wpack,
    const float* __restrict__ bias1, const float* __restrict__ bias2, int rows1,
    float* __restrict__ out, int OC) {
  const int lane   = threadIdx.x & 31;
  const int wave   = threadIdx.x >> 5;
  const int octile = blockIdx.y * 4 + wave;
  const int n      = blockIdx.z;
  const int half   = lane >> 4;

  const int col0 = blockIdx.x * 32 + (lane & 15);
  const int col1 = col0 + 16;
  const bool cv0 = (col0 < 7500);
  const bool cv1 = (col1 < 7500);
  const int cc0 = cv0 ? col0 : 7499;
  const int cc1 = cv1 ? col1 : 7499;
  const int t0 = cc0 / 25, v0 = cc0 - t0 * 25;
  const int t1 = cc1 / 25, v1 = cc1 - t1 * 25;

  const v16bf* wfrag = reinterpret_cast<const v16bf*>(wpack);
  const size_t nbase = (size_t)n * 300;
  v8f acc0 = {};
  v8f acc1 = {};
  for (int k = 0; k < 9; ++k) {
    const int tp0 = t0 + k - 4;
    const int tp1 = t1 + k - 4;
    const bool ok0 = cv0 && ((unsigned)tp0 < 300u);
    const bool ok1 = cv1 && ((unsigned)tp1 < 300u);
    const unsigned short* bp0 =
        xT + ((nbase + (ok0 ? tp0 : 0)) * 25 + v0) * 64 + half * 16;
    const unsigned short* bp1 =
        xT + ((nbase + (ok1 ? tp1 : 0)) * 25 + v1) * 64 + half * 16;
#pragma unroll
    for (int kc = 0; kc < 2; ++kc) {
      v16bf a = wfrag[((octile * 9 + k) * 2 + kc) * 32 + lane];
      v16bf b0 = {};
      v16bf b1 = {};
      if (ok0) b0 = *reinterpret_cast<const v16bf*>(bp0 + kc * 32);
      if (ok1) b1 = *reinterpret_cast<const v16bf*>(bp1 + kc * 32);
      acc0 = __builtin_amdgcn_wmma_f32_16x16x32_bf16(
          false, a, false, b0, (short)0, acc0, false, false);
      acc1 = __builtin_amdgcn_wmma_f32_16x16x32_bf16(
          false, a, false, b1, (short)0, acc1, false, false);
    }
  }
  // C/D layout: VGPR m, lanes<16 -> M=m, lanes>=16 -> M=m+8
#pragma unroll
  for (int m = 0; m < 8; ++m) {
    int row = octile * 16 + half * 8 + m;
    float bv = (row < rows1) ? bias1[row] : bias2[row - rows1];
    float* orow = out + ((size_t)n * OC + row) * 7500;
    if (cv0) orow[col0] = acc0[m] + bv;
    if (cv1) orow[col1] = acc1[m] + bv;
  }
}

// ---------------------------------------------------------------------------
// Attention: logits[i][j] = (1/9600) * sum_{c,t} a1[c,t,i]*a2[c,t,j]
// then softmax over rows i (axis -2).  a12: (N, 64, 7500), ch 0-31 = conv1.
// grid = N, block = 256 (each thread owns up to 3 of the 625 (i,j) pairs)
// ---------------------------------------------------------------------------
__global__ void __launch_bounds__(256) attn_kernel(const float* __restrict__ a12,
                                                   float* __restrict__ att) {
  const int n = blockIdx.x;
  const int tid = threadIdx.x;
  __shared__ float la1[32][26];
  __shared__ float la2[32][26];
  __shared__ float sl[625];
  __shared__ float cmax[25];
  __shared__ float cinv[25];
  const float* base = a12 + (size_t)n * 64 * 7500;
  float acc0 = 0.f, acc1 = 0.f, acc2 = 0.f;
  const int p0 = tid, p1 = tid + 256, p2 = tid + 512;
  const int q0 = (p0 < 625) ? p0 : 0;
  const int q1 = (p1 < 625) ? p1 : 0;
  const int q2 = (p2 < 625) ? p2 : 0;
  const int i0 = q0 / 25, j0 = q0 % 25;
  const int i1 = q1 / 25, j1 = q1 % 25;
  const int i2 = q2 / 25, j2 = q2 % 25;
  for (int chunk = 0; chunk < 300; ++chunk) {   // 9600 (c,t) rows, 32 per chunk
    __syncthreads();
    for (int e = tid; e < 1600; e += 256) {
      int which = (e >= 800);
      int ee = which ? (e - 800) : e;
      int r = ee / 25, i = ee - r * 25;
      int ct = chunk * 32 + r;
      int c = ct / 300, tt = ct - c * 300;
      float val = base[(size_t)(c + (which ? 32 : 0)) * 7500 + tt * 25 + i];
      if (which) la2[r][i] = val; else la1[r][i] = val;
    }
    __syncthreads();
#pragma unroll 8
    for (int r = 0; r < 32; ++r) {
      acc0 += la1[r][i0] * la2[r][j0];
      acc1 += la1[r][i1] * la2[r][j1];
      acc2 += la1[r][i2] * la2[r][j2];
    }
  }
  __syncthreads();
  const float scale = 1.0f / 9600.0f;
  if (p0 < 625) sl[p0] = acc0 * scale;
  if (p1 < 625) sl[p1] = acc1 * scale;
  if (p2 < 625) sl[p2] = acc2 * scale;
  __syncthreads();
  if (tid < 25) {                               // softmax over rows, per column
    float m = -3.402823466e+38f;
    for (int i = 0; i < 25; ++i) m = fmaxf(m, sl[i * 25 + tid]);
    float s = 0.f;
    for (int i = 0; i < 25; ++i) s += expf(sl[i * 25 + tid] - m);
    cmax[tid] = m; cinv[tid] = 1.0f / s;
  }
  __syncthreads();
  float* ao = att + n * 625;
  if (p0 < 625) ao[p0] = expf(sl[p0] - cmax[j0]) * cinv[j0];
  if (p1 < 625) ao[p1] = expf(sl[p1] - cmax[j1]) * cinv[j1];
  if (p2 < 625) ao[p2] = expf(sl[p2] - cmax[j2]) * cinv[j2];
}

// ---------------------------------------------------------------------------
// y1[n,c,t,v'] = sum_v x[n,c,t,v] * att[n,v,v'] ; written channel-last bf16
// (same layout as xT) so the final WMMA conv can consume it directly.
// grid = (75 t-groups of 4, N), block = 256 = 64 channels x 4 t
// ---------------------------------------------------------------------------
__global__ void __launch_bounds__(256) apply_att_kernel(const float* __restrict__ x,
                                                        const float* __restrict__ att,
                                                        unsigned short* __restrict__ y1T) {
  const int n = blockIdx.y;
  const int c = threadIdx.x & 63;
  const int t = blockIdx.x * 4 + (threadIdx.x >> 6);
  __shared__ float satt[625];
  for (int e = threadIdx.x; e < 625; e += 256) satt[e] = att[n * 625 + e];
  __syncthreads();
  float xr[25];
  const float* xp = x + (((size_t)n * 64 + c) * 300 + t) * 25;
#pragma unroll
  for (int v = 0; v < 25; ++v) xr[v] = xp[v];
  unsigned short* op = y1T + ((size_t)n * 300 + t) * 25 * 64 + c;
#pragma unroll
  for (int vp = 0; vp < 25; ++vp) {
    float s = 0.f;
#pragma unroll
    for (int v = 0; v < 25; ++v) s += xr[v] * satt[v * 25 + vp];
    op[(size_t)vp * 64] = f2bf(s);
  }
}

// ---------------------------------------------------------------------------
// Deterministic BN stats: per-(channel,n) partial sum / sumsq (tree reduce).
// ---------------------------------------------------------------------------
__global__ void __launch_bounds__(256) bn_stats_kernel(const float* __restrict__ y,
                                                       float* __restrict__ partial) {
  const int c = blockIdx.x, n = blockIdx.y;
  const float* p = y + ((size_t)n * 128 + c) * 7500;
  float s = 0.f, s2 = 0.f;
  for (int i = threadIdx.x; i < 7500; i += 256) { float v = p[i]; s += v; s2 += v * v; }
  __shared__ float rs[256], rq[256];
  rs[threadIdx.x] = s; rq[threadIdx.x] = s2;
  __syncthreads();
  for (int st = 128; st > 0; st >>= 1) {
    if (threadIdx.x < st) { rs[threadIdx.x] += rs[threadIdx.x + st];
                            rq[threadIdx.x] += rq[threadIdx.x + st]; }
    __syncthreads();
  }
  if (threadIdx.x == 0) { partial[c * 64 + n] = rs[0]; partial[8192 + c * 64 + n] = rq[0]; }
}

__global__ void bn_finalize_kernel(const float* __restrict__ partial,
                                   const float* __restrict__ bn_w,
                                   const float* __restrict__ bn_b,
                                   float* __restrict__ ss) {
  const int c = threadIdx.x;
  if (c >= 128) return;
  float s = 0.f, s2 = 0.f;
  for (int n = 0; n < 64; ++n) { s += partial[c * 64 + n]; s2 += partial[8192 + c * 64 + n]; }
  const float invc = 1.0f / 480000.0f;           // N*T*V
  float mean = s * invc;
  float var  = s2 * invc - mean * mean;          // biased, matches jnp.var
  float r    = rsqrtf(var + 1e-5f);
  float sc   = bn_w[c] * r;
  ss[c]       = sc;
  ss[128 + c] = bn_b[c] - mean * sc;
}

// grid = (30, 128 channels, 64 n), in-place on d_out
__global__ void __launch_bounds__(256) bn_apply_kernel(float* __restrict__ y,
                                                       const float* __restrict__ ss) {
  const int c = blockIdx.y;
  const int i = blockIdx.x * 256 + threadIdx.x;
  if (i >= 7500) return;
  const size_t idx = ((size_t)blockIdx.z * 128 + c) * 7500 + i;
  y[idx] = y[idx] * ss[c] + ss[128 + c];
}

// ---------------------------------------------------------------------------
extern "C" void kernel_launch(void* const* d_in, const int* in_sizes, int n_in,
                              void* d_out, int out_size, void* d_ws, size_t ws_size,
                              hipStream_t stream) {
  (void)in_sizes; (void)n_in; (void)out_size; (void)ws_size;
  const float* x       = (const float*)d_in[0];
  const float* conv_w  = (const float*)d_in[1];
  const float* conv_b  = (const float*)d_in[2];
  const float* conv1_w = (const float*)d_in[3];
  const float* conv1_b = (const float*)d_in[4];
  const float* conv2_w = (const float*)d_in[5];
  const float* conv2_b = (const float*)d_in[6];
  const float* bn_w    = (const float*)d_in[7];
  const float* bn_b    = (const float*)d_in[8];
  float* out = (float*)d_out;

  char* ws = (char*)d_ws;
  size_t off = 0;
  auto alloc = [&](size_t bytes) -> char* {
    char* p = ws + off;
    off += (bytes + 255) & ~(size_t)255;
    return p;
  };
  unsigned short* xT      = (unsigned short*)alloc((size_t)64 * 7500 * 64 * 2); // 61.44 MB
  unsigned short* y1T     = (unsigned short*)alloc((size_t)64 * 7500 * 64 * 2); // 61.44 MB
  float*          a12     = (float*)alloc((size_t)64 * 64 * 7500 * 4);          // 122.9 MB
  float*          att     = (float*)alloc((size_t)64 * 625 * 4);
  unsigned short* wp12    = (unsigned short*)alloc((size_t)64 * 576 * 2);
  unsigned short* wpC     = (unsigned short*)alloc((size_t)128 * 576 * 2);
  float*          partial = (float*)alloc((size_t)2 * 128 * 64 * 4);
  float*          ss      = (float*)alloc((size_t)256 * 4);

  // 1) repack x -> channel-last bf16
  pack_x_kernel<<<dim3(125, 64), 256, 0, stream>>>(x, xT);
  // 2) repack weights into WMMA A-fragment order (conv1+conv2 fused, then conv)
  pack_w_kernel<<<(64 * 576 + 255) / 256, 256, 0, stream>>>(conv1_w, conv2_w, 32, 64, wp12);
  pack_w_kernel<<<(128 * 576 + 255) / 256, 256, 0, stream>>>(conv_w, conv_w, 128, 128, wpC);
  // 3) fused conv1+conv2 (OC=64) via bf16 WMMA -> a12 (N,64,7500)
  tconv_wmma_kernel<<<dim3(235, 1, 64), 128, 0, stream>>>(xT, wp12, conv1_b, conv2_b, 32, a12, 64);
  // 4) 25x25 attention + column softmax
  attn_kernel<<<64, 256, 0, stream>>>(a12, att);
  // 5) x @ att, fused with NHWC+bf16 repack for the final conv
  apply_att_kernel<<<dim3(75, 64), 256, 0, stream>>>(x, att, y1T);
  // 6) final conv (OC=128) via bf16 WMMA -> d_out (pre-BN, N,128,T,V)
  tconv_wmma_kernel<<<dim3(235, 2, 64), 128, 0, stream>>>(y1T, wpC, conv_b, conv_b, 128, out, 128);
  // 7) batchnorm: deterministic two-stage stats, then in-place scale/shift
  bn_stats_kernel<<<dim3(128, 64), 256, 0, stream>>>(out, partial);
  bn_finalize_kernel<<<1, 128, 0, stream>>>(partial, bn_w, bn_b, ss);
  bn_apply_kernel<<<dim3(30, 128, 64), 256, 0, stream>>>(out, ss);
}
